// LandmarkModel_6674379178302
// MI455X (gfx1250) — compile-verified, matrix operand order checked by reading, not run on hardware
//
#include <hip/hip_runtime.h>
#include <hip/hip_bf16.h>

// ---------------------------------------------------------------------------
// Problem constants (from reference): B=32, C=98, H=W=128, F=256, D=128
// ---------------------------------------------------------------------------
#define BB      32
#define CC      98
#define HH      128
#define WW      128
#define HW      (HH * WW)          // 16384
#define FF      256
#define DD      128
#define BC      (BB * CC)          // 3136 rows (multiple of 16)
#define KIN     259                // 2 + 1 + F
#define KPAD    288                // 259 padded to multiple of 32
#define CPAD    112                // 98 padded to multiple of 16
#define HM_ELEMS ((long)BB * CC * HW)   // 51,380,224

typedef __bf16 bf16_t;
typedef __attribute__((ext_vector_type(16))) __bf16 v16bf;
typedef __attribute__((ext_vector_type(8)))  float  v8f;
typedef __attribute__((ext_vector_type(4)))  float  f32x4;
typedef __attribute__((ext_vector_type(4)))  unsigned int u32x4;

__device__ __forceinline__ bf16_t f2bf(float f) {
    union { float f; unsigned u; } x; x.f = f;
    unsigned r = x.u + 0x7FFFu + ((x.u >> 16) & 1u);   // round-to-nearest-even
    unsigned short h = (unsigned short)(r >> 16);
    return __builtin_bit_cast(bf16_t, h);
}

// ---------------------------------------------------------------------------
// K1: streaming copy of hm -> d_out fused with per-(b,c) argmax + max value.
// One block per (b,c); 256 threads; each thread moves 16 float4s (64 KB/blk).
// Nontemporal: hm is 196MB, read once, never reused -> keep out of L2.
// ---------------------------------------------------------------------------
__global__ void argmax_copy_kernel(const float* __restrict__ hm,
                                   float* __restrict__ out_hm,
                                   int* __restrict__ idx_ws,
                                   float* __restrict__ conf_ws) {
    const int  bc   = blockIdx.x;
    const long base = (long)bc * HW;
    const int  t    = threadIdx.x;

    float best = -__builtin_inff();
    int   bidx = 0;
#pragma unroll
    for (int i = 0; i < 16; ++i) {
        const int off = i * 1024 + t * 4;
        f32x4 v = __builtin_nontemporal_load((const f32x4*)(hm + base + off));
        __builtin_nontemporal_store(v, (f32x4*)(out_hm + base + off));
#pragma unroll
        for (int j = 0; j < 4; ++j) {
            const float x = v[j];
            if (x > best) { best = x; bidx = off + j; }   // first-occurrence order
        }
    }

    __shared__ float smax[256];
    __shared__ int   sidx[256];
    smax[t] = best; sidx[t] = bidx;
    __syncthreads();
    for (int s = 128; s > 0; s >>= 1) {
        if (t < s) {
            const float o  = smax[t + s];
            const int   oi = sidx[t + s];
            if (o > smax[t] || (o == smax[t] && oi < sidx[t])) { smax[t] = o; sidx[t] = oi; }
        }
        __syncthreads();
    }
    if (t == 0) { idx_ws[bc] = sidx[0]; conf_ws[bc] = smax[0]; }
}

// ---------------------------------------------------------------------------
// K2: build node matrix [BC, KPAD] in bf16: [x/H, y/W, conf, vis(256), 0-pad]
// One block per (b,c), 288 threads (9 waves). Feature gather is inherently
// scattered (stride 64KB per f), but total traffic is small.
// ---------------------------------------------------------------------------
__global__ void build_node_kernel(const float* __restrict__ feat,
                                  const int* __restrict__ idx_ws,
                                  const float* __restrict__ conf_ws,
                                  bf16_t* __restrict__ node) {
    const int bc  = blockIdx.x;
    const int b   = bc / CC;
    const int t   = threadIdx.x;             // 0..287
    const int idx = idx_ws[bc];

    float val;
    if (t == 0)        val = (float)(idx % WW) * (1.0f / (float)HH);  // xs / hm_size[0]
    else if (t == 1)   val = (float)(idx / WW) * (1.0f / (float)WW);  // ys / hm_size[1]
    else if (t == 2)   val = conf_ws[bc];
    else if (t < KIN)  val = feat[((long)(b * FF + (t - 3))) * HW + idx];
    else               val = 0.0f;
    node[(long)bc * KPAD + t] = f2bf(val);
}

// ---------------------------------------------------------------------------
// Weight prep: bf16 conversion, transposes (so GEMM B-operand is K-contig),
// zero padding.
// ---------------------------------------------------------------------------
__global__ void prep_wembedT(const float* __restrict__ W, bf16_t* __restrict__ WT) {
    const int i = blockIdx.x * blockDim.x + threadIdx.x;     // n*KPAD + k
    if (i >= DD * KPAD) return;
    const int n = i / KPAD, k = i % KPAD;
    WT[i] = (k < KIN) ? f2bf(W[k * DD + n]) : f2bf(0.0f);
}
__global__ void prep_adj(const float* __restrict__ A, bf16_t* __restrict__ Ab) {
    const int i = blockIdx.x * blockDim.x + threadIdx.x;     // r*128 + k
    if (i >= CPAD * 128) return;
    const int r = i / 128, k = i % 128;
    Ab[i] = (r < CC && k < CC) ? f2bf(A[r * CC + k]) : f2bf(0.0f);
}
__global__ void prep_wgcnT(const float* __restrict__ W, bf16_t* __restrict__ WT) {
    const int i = blockIdx.x * blockDim.x + threadIdx.x;     // n*128 + k
    if (i >= DD * DD) return;
    const int n = i / DD, k = i % DD;
    WT[i] = f2bf(W[k * DD + n]);
}
__global__ void zero_bf16_kernel(bf16_t* __restrict__ p, int n) {
    const int i = blockIdx.x * blockDim.x + threadIdx.x;
    if (i < n) p[i] = f2bf(0.0f);
}

// ---------------------------------------------------------------------------
// Generic bf16 WMMA GEMM: D[m,n] = act( sum_k A[m,k]*Bt[n,k] + bias[n] ).
// One wave per 16x16 output tile; grid.z = batch.
// A : row-major [M, lda] bf16 (K contiguous)
// Bt: col-major-as-rows [N, ldb] bf16 (K contiguous)
// Fragment layout per CDNA5 ISA 7.12.2 (16-bit A/B, wave32):
//   lane<16 : row/col = lane,    k in [kb, kb+8) U [kb+16, kb+24)
//   lane>=16: row/col = lane-16, k in [kb+8,kb+16) U [kb+24, kb+32)
// C/D layout: lane = n + 16*(m>=8), vgpr v = m&7.
// STORE_MODE: 0 = fp32 row-major, 1 = bf16 row-major,
//             2 = bf16 "T98": h0T[b][n][k] with b=row/98, k=row%98
// ---------------------------------------------------------------------------
template <int STORE_MODE, bool RELU, bool BIAS>
__global__ void gemm_bf16_wmma(const bf16_t* __restrict__ A, long aStrideB, int lda,
                               const bf16_t* __restrict__ Bt, long bStrideB, int ldb,
                               void* __restrict__ C, long cStrideB, int ldc,
                               const float* __restrict__ bias,
                               int Ntiles, int K) {
    const int tile  = blockIdx.x;
    const int batch = blockIdx.z;
    const int m0    = (tile / Ntiles) * 16;
    const int n0    = (tile % Ntiles) * 16;
    const int lane  = threadIdx.x & 31;
    const int half  = lane >> 4;      // 0 or 1
    const int l15   = lane & 15;

    const bf16_t* Ap = A  + batch * aStrideB + (long)(m0 + l15) * lda + half * 8;
    const bf16_t* Bp = Bt + batch * bStrideB + (long)(n0 + l15) * ldb + half * 8;

    v8f acc = {};
    for (int kb = 0; kb < K; kb += 32) {
        v16bf a, b;
        ((u32x4*)&a)[0] = *(const u32x4*)(Ap + kb);
        ((u32x4*)&a)[1] = *(const u32x4*)(Ap + kb + 16);
        ((u32x4*)&b)[0] = *(const u32x4*)(Bp + kb);
        ((u32x4*)&b)[1] = *(const u32x4*)(Bp + kb + 16);
        acc = __builtin_amdgcn_wmma_f32_16x16x32_bf16(
            /*neg_a=*/false, a, /*neg_b=*/false, b,
            /*c_mod=*/(short)0, acc, /*reuse_a=*/false, /*reuse_b=*/false);
    }

    const int   n    = n0 + l15;
    const float bval = BIAS ? bias[n] : 0.0f;
#pragma unroll
    for (int v = 0; v < 8; ++v) {
        float x = acc[v] + bval;
        if (RELU) x = fmaxf(x, 0.0f);
        const int m = m0 + half * 8 + v;
        if (STORE_MODE == 0) {
            ((float*)C)[batch * cStrideB + (long)m * ldc + n] = x;
        } else if (STORE_MODE == 1) {
            ((bf16_t*)C)[batch * cStrideB + (long)m * ldc + n] = f2bf(x);
        } else {  // T98: scatter row m = b*98 + k into per-batch transposed K-major
            const int bb = m / CC;
            const int kk = m - bb * CC;
            ((bf16_t*)C)[(long)bb * (DD * DD) + n * DD + kk] = f2bf(x);
        }
    }
}

// ---------------------------------------------------------------------------
// Final projection: out[b,c,o] = h1[b,c,:] . W_out[:,o] + b_out[o]  (N=2)
// ---------------------------------------------------------------------------
__global__ void out_proj_kernel(const float* __restrict__ h1,
                                const float* __restrict__ W_out,
                                const float* __restrict__ b_out,
                                float* __restrict__ out) {
    const int i = blockIdx.x * blockDim.x + threadIdx.x;
    if (i >= BC * 2) return;
    const int bc = i >> 1, o = i & 1;
    const int b = bc / CC, c = bc % CC;
    const float* row = h1 + ((long)b * CPAD + c) * DD;
    float acc = b_out[o];
#pragma unroll 4
    for (int d = 0; d < DD; ++d) acc += row[d] * W_out[d * 2 + o];
    out[i] = acc;
}

// ---------------------------------------------------------------------------
extern "C" void kernel_launch(void* const* d_in, const int* in_sizes, int n_in,
                              void* d_out, int out_size, void* d_ws, size_t ws_size,
                              hipStream_t stream) {
    const float* hm      = (const float*)d_in[0];
    const float* feat    = (const float*)d_in[1];
    const float* W_embed = (const float*)d_in[2];
    const float* b_embed = (const float*)d_in[3];
    const float* A_adj   = (const float*)d_in[4];
    const float* W_gcn   = (const float*)d_in[5];
    const float* W_out   = (const float*)d_in[6];
    const float* b_out   = (const float*)d_in[7];
    float* out = (float*)d_out;

    // workspace layout (256B aligned)
    char* ws = (char*)d_ws;
    size_t off = 0;
    auto take = [&](size_t bytes) -> char* {
        char* p = ws + off;
        off = (off + bytes + 255) & ~(size_t)255;
        return p;
    };
    int*    idx_ws  = (int*)   take(BC * sizeof(int));
    float*  conf_ws = (float*) take(BC * sizeof(float));
    bf16_t* node_bf = (bf16_t*)take((size_t)BC * KPAD * 2);        // [3136,288]
    bf16_t* WembT   = (bf16_t*)take((size_t)DD * KPAD * 2);        // [128,288]
    bf16_t* Aadj_bf = (bf16_t*)take((size_t)CPAD * 128 * 2);       // [112,128]
    bf16_t* WgcnT   = (bf16_t*)take((size_t)DD * DD * 2);          // [128,128]
    bf16_t* h0T     = (bf16_t*)take((size_t)BB * DD * DD * 2);     // [32][128n][128k]
    bf16_t* t_bf    = (bf16_t*)take((size_t)BB * CPAD * DD * 2);   // [32][112][128]
    float*  h1      = (float*) take((size_t)BB * CPAD * DD * 4);   // [32][112][128]
    (void)ws_size;

    // 1) stream-copy hm -> out[0] + fused argmax/conf
    argmax_copy_kernel<<<BC, 256, 0, stream>>>(hm, out, idx_ws, conf_ws);

    // 2) weight prep (independent of K1 data, same stream ordering is fine)
    prep_wembedT<<<(DD * KPAD + 255) / 256, 256, 0, stream>>>(W_embed, WembT);
    prep_adj   <<<(CPAD * 128 + 255) / 256, 256, 0, stream>>>(A_adj, Aadj_bf);
    prep_wgcnT <<<(DD * DD + 255) / 256, 256, 0, stream>>>(W_gcn, WgcnT);
    zero_bf16_kernel<<<(BB * DD * DD + 255) / 256, 256, 0, stream>>>(h0T, BB * DD * DD);

    // 3) gather features + build padded bf16 node matrix
    build_node_kernel<<<BC, KPAD, 0, stream>>>(feat, idx_ws, conf_ws, node_bf);

    // 4) G1: h0 = relu(node @ W_embed + b_embed)   M=3136 N=128 K=288
    //    stored bf16 transposed per batch -> B-operand of G2
    {
        dim3 grid((BC / 16) * (DD / 16), 1, 1);
        gemm_bf16_wmma<2, true, true><<<grid, 32, 0, stream>>>(
            node_bf, 0, KPAD, WembT, 0, KPAD, h0T, 0, DD, b_embed, DD / 16, KPAD);
    }
    // 5) G2: t[b] = A_adj @ h0[b]   M=112 N=128 K=128, per batch
    {
        dim3 grid((CPAD / 16) * (DD / 16), 1, BB);
        gemm_bf16_wmma<1, false, false><<<grid, 32, 0, stream>>>(
            Aadj_bf, 0, 128, h0T, (long)DD * DD, DD, t_bf, (long)CPAD * DD, DD,
            nullptr, DD / 16, 128);
    }
    // 6) G3: h1[b] = relu(t[b] @ W_gcn)   M=112 N=128 K=128, per batch
    {
        dim3 grid((CPAD / 16) * (DD / 16), 1, BB);
        gemm_bf16_wmma<0, true, false><<<grid, 32, 0, stream>>>(
            t_bf, (long)CPAD * DD, DD, WgcnT, 0, DD, h1, (long)CPAD * DD, DD,
            nullptr, DD / 16, 128);
    }
    // 7) final 128->2 projection into out[1] region
    out_proj_kernel<<<(BC * 2 + 255) / 256, 256, 0, stream>>>(
        h1, W_out, b_out, out + HM_ELEMS);
}
